// WindowMultiHeadSelfAttention_12266426597632
// MI455X (gfx1250) — matrix operand order repaired and checked by reading, not compiled, not run on hardware
//
#include <hip/hip_runtime.h>
#include <hip/hip_bf16.h>

typedef _Float16 half_t;
typedef _Float16 v16h __attribute__((ext_vector_type(16)));
typedef _Float16 v8h  __attribute__((ext_vector_type(8)));
typedef _Float16 v4h  __attribute__((ext_vector_type(4)));
typedef float    v8f  __attribute__((ext_vector_type(8)));

#define DIMX 384
#define NH   12
#define NTOK 49
#define BN   4096

// ---- fragment loaders (layouts per CDNA5 ISA 7.12.2, 16-bit A/B) ----
// A 16x32: lane holds row (l&15); halfs = K[kb..kb+7] , K[kb+16..kb+23], kb=(l<16)?0:8
__device__ __forceinline__ v16h load_a16(const half_t* p) {
    v8h lo = *(const v8h*)(p);
    v8h hi = *(const v8h*)(p + 16);
    return __builtin_shufflevector(lo, hi, 0,1,2,3,4,5,6,7,8,9,10,11,12,13,14,15);
}
// B 32x16: lane holds col (l&15); halfs = K[kb..kb+7] , K[kb+16..kb+23], kb=(l<16)?0:8
__device__ __forceinline__ v16h load_b16(const half_t* p) {
    v8h lo = *(const v8h*)(p);
    v8h hi = *(const v8h*)(p + 16);
    return __builtin_shufflevector(lo, hi, 0,1,2,3,4,5,6,7,8,9,10,11,12,13,14,15);
}
__device__ __forceinline__ v8f wmma16(v16h a, v16h b, v8f c) {
    return __builtin_amdgcn_wmma_f32_16x16x32_f16(false, a, false, b, (short)0, c, false, false);
}

// ---- pre-pass: fp32 -> f16 weights (stay hot in 192MB L2) ----
__global__ void cvt_weights_kernel(const float* __restrict__ wqkv,
                                   const float* __restrict__ wproj,
                                   half_t* __restrict__ wqkv_h,
                                   half_t* __restrict__ wproj_h) {
    int i = blockIdx.x * 256 + threadIdx.x;
    if (i < 3 * DIMX * DIMX) wqkv_h[i] = (half_t)wqkv[i];
    if (i < DIMX * DIMX)     wproj_h[i] = (half_t)wproj[i];
}

// ---- pre-pass: gather relative position bias -> [12][49][49] f32 ----
__global__ void rel_bias_kernel(const float* __restrict__ table, float* __restrict__ relb) {
    int i = blockIdx.x * 256 + threadIdx.x;
    if (i >= NH * NTOK * NTOK) return;
    int h = i / (NTOK * NTOK);
    int rem = i % (NTOK * NTOK);
    int a = rem / NTOK, b = rem % NTOK;
    int idx = (a / 7 - b / 7 + 6) * 13 + (a % 7 - b % 7 + 6);
    relb[i] = table[idx * NH + h];
}

// LDS layout (bytes):
//  [0      , 65536 ) : stage0/1: xh f16 [64][384]; stage2: per-wave probs f16 [8][64][64]
//  [65536  , 114688) : q  f16 [12][64][32]
//  [114688 , 163840) : k  f16 [12][64][32]
//  [163840 , 212992) : vT f16 [12][32][64]   (zero-padded cols 49..63)
//  [212992 , 262144) : ctx f16 [64][384]
#define SMEM_BYTES 262144

__global__ void __launch_bounds__(256)
swin_window_attn_kernel(const float* __restrict__ x, const float* __restrict__ mask,
                        const float* __restrict__ b_qkv, const float* __restrict__ b_proj,
                        const half_t* __restrict__ wqkv_h, const half_t* __restrict__ wproj_h,
                        const float* __restrict__ relb, float* __restrict__ out) {
    extern __shared__ char smem[];
    half_t* sh  = (half_t*)smem;
    half_t* xh  = sh;             // 64*384
    half_t* qsm = sh + 32768;     // 12*64*32
    half_t* ksm = sh + 57344;
    half_t* vsm = sh + 81920;     // 12*32*64 (transposed V)
    half_t* csm = sh + 106496;    // 64*384

    const int tid  = threadIdx.x;
    const int lane = tid & 31;
    const int wid  = tid >> 5;
    const int ln16 = lane & 15;
    const bool hih = lane >= 16;
    const int kbA = hih ? 8 : 0;
    const int kbB = hih ? 8 : 0;
    const int bn = blockIdx.x;

    // ---------- stage 0: x -> f16 LDS (vectorized, zero-padded to 64 rows); zero vT ----------
    {
        const float4* xv = (const float4*)(x + (size_t)bn * NTOK * DIMX);
        for (int i = tid; i < NTOK * DIMX / 4; i += 256) {
            float4 f = xv[i];
            v4h h; h[0] = (half_t)f.x; h[1] = (half_t)f.y; h[2] = (half_t)f.z; h[3] = (half_t)f.w;
            *(v4h*)(xh + i * 4) = h;
        }
        v4h z; z[0] = z[1] = z[2] = z[3] = (half_t)0.f;
        for (int i = tid; i < (64 - NTOK) * DIMX / 4; i += 256)
            *(v4h*)(xh + NTOK * DIMX + i * 4) = z;
        for (int i = tid; i < NH * 32 * 64 / 4; i += 256)
            *(v4h*)(vsm + i * 4) = z;
    }
    __syncthreads();

    // ---------- stage 1: QKV = x @ Wqkv^T + b ; scatter to q/k/vT LDS ----------
    // mtile outer: 12 A fragments cached in VGPRs, reused across this wave's 9 N-tiles.
    for (int mtile = 0; mtile < 4; ++mtile) {
        v16h areg[12];
        {
            const half_t* aP = xh + (mtile * 16 + ln16) * DIMX + kbA;
#pragma unroll
            for (int ks = 0; ks < 12; ++ks) areg[ks] = load_a16(aP + ks * 32);
        }
        for (int ntile = wid; ntile < 72; ntile += 8) {
            const int n = ntile * 16 + ln16;
            v8f c; float bv = b_qkv[n];
#pragma unroll
            for (int r = 0; r < 8; ++r) c[r] = bv;
            const half_t* bP = wqkv_h + (size_t)n * DIMX + kbB;
#pragma unroll
            for (int ks = 0; ks < 12; ++ks)
                c = wmma16(areg[ks], load_b16(bP + ks * 32), c);
            const int part = n / DIMX, rem = n % DIMX;
            const int head = rem >> 5, dd = rem & 31;
#pragma unroll
            for (int r = 0; r < 8; ++r) {
                int m = mtile * 16 + (hih ? 8 + r : r);
                half_t hv = (half_t)c[r];
                if (part == 0)      qsm[head * 2048 + m * 32 + dd] = hv;
                else if (part == 1) ksm[head * 2048 + m * 32 + dd] = hv;
                else if (m < NTOK)  vsm[head * 2048 + dd * 64 + m] = hv;  // transposed, pad stays 0
            }
        }
    }
    __syncthreads();

    // ---------- stage 2: per-head attention ----------
    const float scale = 0.1767766953f;   // 32^-0.5
    const int win = bn >> 6;             // bn / B
    half_t* probsW = sh + wid * 4096;    // per-wave [64][64] f16 scratch (aliases xh region)
    for (int h = wid; h < NH; h += 8) {
        const half_t* qh = qsm + h * 2048;
        const half_t* kh = ksm + h * 2048;
        const half_t* vh = vsm + h * 2048;
        v16h kf[4];
#pragma unroll
        for (int nt = 0; nt < 4; ++nt)
            kf[nt] = load_b16(kh + (nt * 16 + ln16) * 32 + kbB);
        const float* rbh = relb + h * (NTOK * NTOK);
        const float* mkw = mask + (size_t)win * NTOK * NTOK;

        for (int mtile = 0; mtile < 4; ++mtile) {
            v16h a = load_a16(qh + (mtile * 16 + ln16) * 32 + kbA);
            v8f s[4];
#pragma unroll
            for (int nt = 0; nt < 4; ++nt) {
                v8f z;
#pragma unroll
                for (int r = 0; r < 8; ++r) z[r] = 0.f;
                s[nt] = wmma16(a, kf[nt], z);      // QK^T tile
            }
            // scale + bias + mask + row softmax (rows per half-wave, cols across 16 lanes)
            const int rbase = mtile * 16 + (hih ? 8 : 0);
            float p[4][8];
#pragma unroll
            for (int r = 0; r < 8; ++r) {
                int i = rbase + r;
                float mx = -3.0e38f;
#pragma unroll
                for (int nt = 0; nt < 4; ++nt) {
                    int j = nt * 16 + ln16;
                    float sv = -3.0e38f;
                    if (i < NTOK && j < NTOK)
                        sv = s[nt][r] * scale + rbh[i * NTOK + j] + mkw[i * NTOK + j];
                    p[nt][r] = sv;
                    mx = fmaxf(mx, sv);
                }
                for (int off = 8; off; off >>= 1) mx = fmaxf(mx, __shfl_xor(mx, off, 32));
                float sum = 0.f;
#pragma unroll
                for (int nt = 0; nt < 4; ++nt) {
                    int j = nt * 16 + ln16;
                    float e = (i < NTOK && j < NTOK) ? __expf(p[nt][r] - mx) : 0.f;
                    p[nt][r] = e; sum += e;
                }
                for (int off = 8; off; off >>= 1) sum += __shfl_xor(sum, off, 32);
                float inv = 1.0f / fmaxf(sum, 1e-20f);
#pragma unroll
                for (int nt = 0; nt < 4; ++nt) p[nt][r] *= inv;
            }
#pragma unroll
            for (int r = 0; r < 8; ++r) {
                int i = rbase + r;
#pragma unroll
                for (int nt = 0; nt < 4; ++nt)
                    probsW[i * 64 + nt * 16 + ln16] = (half_t)p[nt][r];
            }
        }
        // ctx = probs @ V  (K padded to 64; pad cols of probs & vT are exact zeros)
        for (int mtile = 0; mtile < 4; ++mtile) {
            const half_t* aP = probsW + (mtile * 16 + ln16) * 64 + kbA;
            v16h a0 = load_a16(aP);
            v16h a1 = load_a16(aP + 32);
#pragma unroll
            for (int nt = 0; nt < 2; ++nt) {
                const int d = nt * 16 + ln16;
                v8f c;
#pragma unroll
                for (int r = 0; r < 8; ++r) c[r] = 0.f;
                c = wmma16(a0, load_b16(vh + d * 64 + kbB), c);
                c = wmma16(a1, load_b16(vh + d * 64 + 32 + kbB), c);
#pragma unroll
                for (int r = 0; r < 8; ++r) {
                    int m = mtile * 16 + (hih ? 8 + r : r);
                    csm[m * DIMX + h * 32 + d] = (half_t)c[r];
                }
            }
        }
    }
    __syncthreads();

    // ---------- stage 3: out = ctx @ Wproj^T + b (A cached per mtile, reused over 3 N-tiles) ----------
    float* orow = out + (size_t)bn * NTOK * DIMX;
    for (int mtile = 0; mtile < 4; ++mtile) {
        v16h areg[12];
        {
            const half_t* aP = csm + (mtile * 16 + ln16) * DIMX + kbA;
#pragma unroll
            for (int ks = 0; ks < 12; ++ks) areg[ks] = load_a16(aP + ks * 32);
        }
        for (int ntile = wid; ntile < 24; ntile += 8) {
            const int n = ntile * 16 + ln16;
            v8f c; float bv = b_proj[n];
#pragma unroll
            for (int r = 0; r < 8; ++r) c[r] = bv;
            const half_t* bP = wproj_h + (size_t)n * DIMX + kbB;
#pragma unroll
            for (int ks = 0; ks < 12; ++ks)
                c = wmma16(areg[ks], load_b16(bP + ks * 32), c);
#pragma unroll
            for (int r = 0; r < 8; ++r) {
                int m = mtile * 16 + (hih ? 8 + r : r);
                if (m < NTOK) orow[(size_t)m * DIMX + n] = c[r];
            }
        }
    }
}

extern "C" void kernel_launch(void* const* d_in, const int* in_sizes, int n_in,
                              void* d_out, int out_size, void* d_ws, size_t ws_size,
                              hipStream_t stream) {
    (void)in_sizes; (void)n_in; (void)out_size; (void)ws_size;
    const float* x     = (const float*)d_in[0];
    const float* mask  = (const float*)d_in[1];
    const float* wqkv  = (const float*)d_in[2];
    const float* bqkv  = (const float*)d_in[3];
    const float* wproj = (const float*)d_in[4];
    const float* bproj = (const float*)d_in[5];
    const float* table = (const float*)d_in[6];
    float* out = (float*)d_out;

    // workspace: f16 Wqkv (884736 B) | f16 Wproj (294912 B) | rel bias f32 (115248 B)
    half_t* wqkv_h  = (half_t*)d_ws;
    half_t* wproj_h = wqkv_h + 3 * DIMX * DIMX;
    float*  relb    = (float*)((char*)d_ws + (3 * DIMX * DIMX + DIMX * DIMX) * sizeof(half_t));

    cvt_weights_kernel<<<(3 * DIMX * DIMX + 255) / 256, 256, 0, stream>>>(wqkv, wproj, wqkv_h, wproj_h);
    rel_bias_kernel<<<(NH * NTOK * NTOK + 255) / 256, 256, 0, stream>>>(table, relb);
    swin_window_attn_kernel<<<BN, 256, SMEM_BYTES, stream>>>(x, mask, bqkv, bproj,
                                                             wqkv_h, wproj_h, relb, out);
}
